// KPConv_52312701665288
// MI455X (gfx1250) — matrix operand order repaired and checked by reading, not compiled, not run on hardware
//
#include <hip/hip_runtime.h>

// ---------------- problem constants ----------------
#define BB   4
#define NN   16384
#define SS   16384
#define KK   32
#define CI   64
#define CO   64
#define LL   27          // kernel points
#define LP   32          // padded L

#define KP_SCL     0.57735026919f   // RADIUS / sqrt(3): kernel point coord scale
#define INV_EXT    1.15470053838f   // 1 / KP_EXTENT, KP_EXTENT = 0.5*sqrt(3)

// ---------------- WMMA types ----------------
typedef _Float16 v16h __attribute__((ext_vector_type(16)));
typedef _Float16 v8h  __attribute__((ext_vector_type(8)));
typedef _Float16 v2h  __attribute__((ext_vector_type(2)));
typedef float    v8f  __attribute__((ext_vector_type(8)));

union Frag { v16h f; v8h h[2]; };

// ---------------- LDS layout ----------------
// wb[p][l][k]  : correlation weights, f16, p<16, l<32 (27..31 zero), k<32
//                pitch: l row = 64B, point = 2048B                  -> 32 KB
// xg[p][i][k]  : gathered features f16, channel-major, pitch_i=80B  -> 80 KB
//                (80B = 16B aligned; bank stride 20 -> conflict-light)
//                After stage B, H[p][l*64+i] f16 (4 KB + p*16 skew) is
//                written over the same per-point 5120B region.
#define WB_OFF   0
#define WB_PT    2048
#define XG_OFF   32768
#define XG_PT    5120
#define XG_PITCH 80
#define LDS_BYTES (32768 + 16*5120)   // 114688

// ---------------- weight pre-swizzle ----------------
// Stage-C B operand fragments stored in exact register order:
// wsw[((chunk*4 + cot)*32 + lane)*16 + j],  kk = chunk*32 + (lane>>4)*16 + j,
// o = cot*16 + (lane&15), value = (kk>>6)<27 ? W[kk>>6][kk&63][o] : 0
__global__ void kpconv_prep_weights(const float* __restrict__ w,
                                    _Float16* __restrict__ wsw) {
    int id    = blockIdx.x * 256 + threadIdx.x;     // 131072 total
    int j     = id & 15;
    int lane  = (id >> 4) & 31;
    int cot   = (id >> 9) & 3;
    int chunk = id >> 11;
    int kk = chunk * 32 + ((lane >> 4) * 16) + j;
    int o  = cot * 16 + (lane & 15);
    int l  = kk >> 6, ci = kk & 63;
    float v = (l < LL) ? w[(l * CI + ci) * CO + o] : 0.0f;
    wsw[id] = (_Float16)v;
}

// ---------------- main kernel ----------------
__global__ void kpconv_main(const float* __restrict__ q_pts,
                            const float* __restrict__ s_pts,
                            const int*   __restrict__ inds,
                            const float* __restrict__ x,
                            const _Float16* __restrict__ wsw,
                            float* __restrict__ out) {
    extern __shared__ char smem[];

    const int b  = blockIdx.x / (NN / 16);
    const int n0 = (blockIdx.x % (NN / 16)) * 16;
    const int tid = threadIdx.x;

    // ---------------- Phase 1: gather + correlation weights ----------------
    for (int t = tid; t < 256; t += 128) {
        const int p  = t >> 4;
        const int kp = t & 15;
        const int k0 = kp << 1;
        const int n  = n0 + p;

        const float* q = q_pts + (size_t)(b * NN + n) * 3;
        const float qx = q[0], qy = q[1], qz = q[2];

        const int ib = (b * NN + n) * KK + k0;
        const int i0 = inds[ib], i1 = inds[ib + 1];
        const bool v0 = (i0 < SS), v1 = (i1 < SS);
        const int c0i = v0 ? i0 : 0, c1i = v1 ? i1 : 0;

        const float* sp0 = s_pts + (size_t)(b * SS + c0i) * 3;
        const float* sp1 = s_pts + (size_t)(b * SS + c1i) * 3;
        const float d0x = (v0 ? sp0[0] : 1e6f) - qx;
        const float d0y = (v0 ? sp0[1] : 1e6f) - qy;
        const float d0z = (v0 ? sp0[2] : 1e6f) - qz;
        const float d1x = (v1 ? sp1[0] : 1e6f) - qx;
        const float d1y = (v1 ? sp1[1] : 1e6f) - qy;
        const float d1z = (v1 ? sp1[2] : 1e6f) - qz;

        // ||d - s*e||^2 = ||d||^2 - 2s*(e.d) + s^2*||e||^2,  e in {-1,0,1}^3
        const float dd0 = fmaf(d0x, d0x, fmaf(d0y, d0y, d0z * d0z));
        const float dd1 = fmaf(d1x, d1x, fmaf(d1y, d1y, d1z * d1z));
        const float m2s = -2.0f * KP_SCL;
        const float s2  = KP_SCL * KP_SCL;

        char* wrow = smem + WB_OFF + p * WB_PT + k0 * 2;
        #pragma unroll
        for (int j1 = 0; j1 < 3; ++j1) {          // e1 pairs with d.y (c1 = g[l/9])
            const float e1 = (float)(j1 - 1);
            #pragma unroll
            for (int j0 = 0; j0 < 3; ++j0) {      // e0 pairs with d.x (c0 = g[(l/3)%3])
                const float e0 = (float)(j0 - 1);
                const float u0 = e0 * d0x + e1 * d0y;
                const float u1 = e0 * d1x + e1 * d1y;
                const int nz01 = (j0 != 1) + (j1 != 1);
                #pragma unroll
                for (int j2 = 0; j2 < 3; ++j2) {  // e2 pairs with d.z (c2 = g[l%3])
                    const float e2 = (float)(j2 - 1);
                    const int l   = j1 * 9 + j0 * 3 + j2;
                    const float cc = s2 * (float)(nz01 + (j2 != 1));
                    const float t0 = u0 + e2 * d0z;
                    const float t1 = u1 + e2 * d1z;
                    const float sq0 = fmaf(m2s, t0, dd0 + cc);
                    const float sq1 = fmaf(m2s, t1, dd1 + cc);
                    const float w0 = fmaxf(0.0f, 1.0f - sqrtf(sq0) * INV_EXT);
                    const float w1 = fmaxf(0.0f, 1.0f - sqrtf(sq1) * INV_EXT);
                    v2h pw; pw.x = (_Float16)w0; pw.y = (_Float16)w1;
                    *(v2h*)(wrow + l * 64) = pw;
                }
            }
        }
        #pragma unroll
        for (int l = LL; l < LP; ++l) {             // zero-pad rows 27..31
            v2h z; z.x = (_Float16)0.0f; z.y = (_Float16)0.0f;
            *(v2h*)(wrow + l * 64) = z;
        }

        // features: two rows, packed (k0,k0+1) per channel, channel-major LDS
        const float4* r0 = (const float4*)(x + (size_t)(b * SS + c0i) * CI);
        const float4* r1 = (const float4*)(x + (size_t)(b * SS + c1i) * CI);
        const float m0 = v0 ? 1.0f : 0.0f;
        const float m1 = v1 ? 1.0f : 0.0f;
        char* xrow = smem + XG_OFF + p * XG_PT + k0 * 2;
        #pragma unroll
        for (int i4 = 0; i4 < 16; ++i4) {
            float4 a = r0[i4], c = r1[i4];
            a.x *= m0; a.y *= m0; a.z *= m0; a.w *= m0;
            c.x *= m1; c.y *= m1; c.z *= m1; c.w *= m1;
            const int i = i4 * 4;
            v2h p0; p0.x = (_Float16)a.x; p0.y = (_Float16)c.x;
            v2h p1; p1.x = (_Float16)a.y; p1.y = (_Float16)c.y;
            v2h p2; p2.x = (_Float16)a.z; p2.y = (_Float16)c.z;
            v2h p3; p3.x = (_Float16)a.w; p3.y = (_Float16)c.w;
            *(v2h*)(xrow + (i + 0) * XG_PITCH) = p0;
            *(v2h*)(xrow + (i + 1) * XG_PITCH) = p1;
            *(v2h*)(xrow + (i + 2) * XG_PITCH) = p2;
            *(v2h*)(xrow + (i + 3) * XG_PITCH) = p3;
        }
    }
    __syncthreads();

    // ---------------- Phase 2: per-point H = wb x xg via WMMA -----------------
    const int wv   = tid >> 5;
    const int lane = tid & 31;
    const int m    = lane & 15;
    const int half = lane >> 4;

    #pragma unroll
    for (int pp = 0; pp < 4; ++pp) {
        const int p = wv * 4 + pp;
        char* wbp = smem + WB_OFF + p * WB_PT;
        // A operand: rows = l, K = k.  lane row m (+16 for mtile 1),
        // slots j<8 -> k = half*8+j ; j>=8 -> k = 16+half*8+(j-8)
        Frag A0, A1;
        A0.h[0] = *(const v8h*)(wbp + (m)      * 64 + half * 16);
        A0.h[1] = *(const v8h*)(wbp + (m)      * 64 + half * 16 + 32);
        A1.h[0] = *(const v8h*)(wbp + (16 + m) * 64 + half * 16);
        A1.h[1] = *(const v8h*)(wbp + (16 + m) * 64 + half * 16 + 32);

        // B operand: rows = k, cols = i. lane col m, K slot j<8 -> k = half*8+j ; j>=8 -> k = 16+half*8+(j-8)
        char* xgp = smem + XG_OFF + p * XG_PT;
        Frag Bf[4];
        #pragma unroll
        for (int nt = 0; nt < 4; ++nt) {
            const int i = nt * 16 + m;
            Bf[nt].h[0] = *(const v8h*)(xgp + i * XG_PITCH + half * 16);
            Bf[nt].h[1] = *(const v8h*)(xgp + i * XG_PITCH + half * 16 + 32);
        }

        v8f acc[2][4];
        const v8f z = {};
        #pragma unroll
        for (int nt = 0; nt < 4; ++nt) {
            acc[0][nt] = __builtin_amdgcn_wmma_f32_16x16x32_f16(
                false, A0.f, false, Bf[nt].f, (short)0, z, false, false);
            acc[1][nt] = __builtin_amdgcn_wmma_f32_16x16x32_f16(
                false, A1.f, false, Bf[nt].f, (short)0, z, false, false);
        }

        // write H (f16) over the xg region, per-point 16B skew for bank spread
        char* hb = smem + XG_OFF + p * XG_PT + p * 16;
        #pragma unroll
        for (int mt = 0; mt < 2; ++mt)
            #pragma unroll
            for (int nt = 0; nt < 4; ++nt)
                #pragma unroll
                for (int r = 0; r < 8; ++r) {
                    const int l = mt * 16 + half * 8 + r;
                    const int i = nt * 16 + m;
                    *(_Float16*)(hb + (l * 64 + i) * 2) =
                        (_Float16)acc[mt][nt][r];
                }
    }
    __syncthreads();

    // ---------------- Phase 3: out = H_flat[16 x 2048] @ Wsw[2048 x 64] -------
    v8f acc = {};
    char* hl = smem + XG_OFF + m * XG_PT + m * 16;   // A row = point m
    const v8h* wg = (const v8h*)wsw;
    #pragma unroll 4
    for (int chunk = 0; chunk < 64; ++chunk) {
        Frag A;
        A.h[0] = *(const v8h*)(hl + chunk * 64 + half * 32);
        A.h[1] = *(const v8h*)(hl + chunk * 64 + half * 32 + 16);
        Frag Bw;
        const size_t fo = ((size_t)(chunk * 4 + wv) * 32 + lane) * 2;
        Bw.h[0] = wg[fo];
        Bw.h[1] = wg[fo + 1];
        acc = __builtin_amdgcn_wmma_f32_16x16x32_f16(
            false, A.f, false, Bw.f, (short)0, acc, false, false);
    }

    float* op = out + (size_t)(b * NN + n0) * CO + wv * 16 + m;
    #pragma unroll
    for (int r = 0; r < 8; ++r)
        op[(size_t)(half * 8 + r) * CO] = acc[r];
}

// ---------------- launcher ----------------
extern "C" void kernel_launch(void* const* d_in, const int* in_sizes, int n_in,
                              void* d_out, int out_size, void* d_ws, size_t ws_size,
                              hipStream_t stream) {
    const float* q_pts = (const float*)d_in[0];
    const float* s_pts = (const float*)d_in[1];
    const int*   inds  = (const int*)d_in[2];
    const float* x     = (const float*)d_in[3];
    const float* w     = (const float*)d_in[4];
    float*       out   = (float*)d_out;
    _Float16*    wsw   = (_Float16*)d_ws;   // needs 64*4*32*16*2 = 256 KB

    kpconv_prep_weights<<<512, 256, 0, stream>>>(w, wsw);
    kpconv_main<<<(BB * NN) / 16, 128, LDS_BYTES, stream>>>(
        q_pts, s_pts, inds, x, wsw, out);
}
